// RETAIN_85993835200802
// MI455X (gfx1250) — matrix operand order repaired and checked by reading, not compiled, not run on hardware
//
#include <hip/hip_runtime.h>
#include <hip/hip_bf16.h>
#include <cstdint>

typedef __attribute__((ext_vector_type(16))) _Float16 v16h;
typedef __attribute__((ext_vector_type(8)))  _Float16 h8v;
typedef __attribute__((ext_vector_type(8)))  float    v8f;

// ---------------------------------------------------------------------------
// f32 -> f16 weight conversion
// ---------------------------------------------------------------------------
__global__ void cvt_f32_f16(const float* __restrict__ src, _Float16* __restrict__ dst, int n) {
    int i = blockIdx.x * 256 + threadIdx.x;
    if (i < n) dst[i] = (_Float16)src[i];
}

// ---------------------------------------------------------------------------
// Generic WMMA GEMM:  C[M,N] = act(A[M,K] @ B^T + bias),  B given row-major [N,K] in f16.
// BM=128, BN=64, BK=32, 256 threads (8 waves). M%128==0, N%64==0, K arbitrary (%8==0).
// ACT: 0=none, 1=relu, 2=tanh
// ---------------------------------------------------------------------------
template<int ACT>
__global__ void __launch_bounds__(256)
gemm_wmma(const float* __restrict__ A, const _Float16* __restrict__ B,
          const float* __restrict__ bias, float* __restrict__ C,
          int M, int N, int K, int ldc)
{
    __shared__ _Float16 As[128][32];
    __shared__ _Float16 Bs[64][32];
    const int tid     = threadIdx.x;
    const int lane    = tid & 31;
    const int wave    = tid >> 5;
    const int lane_lo = lane & 15;
    const int khalf   = lane >> 4;
    const int m0 = blockIdx.x * 128;
    const int n0 = blockIdx.y * 64;
    // 4 waves along M, 2 along N; each wave owns a 32x32 quad of 16x16 tiles
    const int wm = (wave >> 1) * 32;
    const int wn = (wave & 1) * 32;

    v8f acc[2][2] = {};
    const int nK = (K + 31) / 32;
    for (int kc = 0; kc < nK; ++kc) {
        const int k0 = kc * 32;
        // stage A (f32 -> f16): 128 rows x 32 k ; thread: row=tid/2, 16 halves
        {
            const int r  = tid >> 1;
            const int kk = (tid & 1) * 16;
            const float* src = A + (size_t)(m0 + r) * K + k0 + kk;
            if (k0 + 32 <= K) {
                #pragma unroll
                for (int i = 0; i < 16; ++i) As[r][kk + i] = (_Float16)src[i];
            } else {
                #pragma unroll
                for (int i = 0; i < 16; ++i)
                    As[r][kk + i] = (k0 + kk + i < K) ? (_Float16)src[i] : (_Float16)0.f;
            }
        }
        // stage B (f16 copy, n-major): 64 rows x 32 k ; thread: row=tid/4, 8 halves
        {
            const int r  = tid >> 2;
            const int kk = (tid & 3) * 8;
            const _Float16* src = B + (size_t)(n0 + r) * K + k0 + kk;
            if (k0 + 32 <= K) {
                #pragma unroll
                for (int i = 0; i < 8; ++i) Bs[r][kk + i] = src[i];
            } else {
                #pragma unroll
                for (int i = 0; i < 8; ++i)
                    Bs[r][kk + i] = (k0 + kk + i < K) ? src[i] : (_Float16)0.f;
            }
        }
        __syncthreads();

        // A fragments: lane=M (lane_lo), VGPR pairs: k = khalf*8+{0..7}, 16+khalf*8+{0..7}
        v16h afr[2];
        #pragma unroll
        for (int i = 0; i < 2; ++i) {
            const int m = wm + i * 16 + lane_lo;
            h8v lo = *(const h8v*)&As[m][khalf * 8];
            h8v hi = *(const h8v*)&As[m][16 + khalf * 8];
            #pragma unroll
            for (int e = 0; e < 8; ++e) { afr[i][e] = lo[e]; afr[i][8 + e] = hi[e]; }
        }
        // B fragments: lane=N (lane_lo), k = khalf*8+{0..7}, 16+khalf*8+{0..7}
        v16h bfr[2];
        #pragma unroll
        for (int j = 0; j < 2; ++j) {
            const int n = wn + j * 16 + lane_lo;
            h8v lo = *(const h8v*)&Bs[n][khalf * 8];
            h8v hi = *(const h8v*)&Bs[n][16 + khalf * 8];
            #pragma unroll
            for (int e = 0; e < 8; ++e) { bfr[j][e] = lo[e]; bfr[j][8 + e] = hi[e]; }
        }
        #pragma unroll
        for (int i = 0; i < 2; ++i)
            #pragma unroll
            for (int j = 0; j < 2; ++j)
                acc[i][j] = __builtin_amdgcn_wmma_f32_16x16x32_f16(
                    false, afr[i], false, bfr[j], (short)0, acc[i][j], false, false);
        __syncthreads();
    }
    // epilogue: C layout f32 16x16 -> row = 8*laneGroup + vgpr, col = lane_lo
    #pragma unroll
    for (int i = 0; i < 2; ++i) {
        #pragma unroll
        for (int j = 0; j < 2; ++j) {
            const int col = n0 + wn + j * 16 + lane_lo;
            const float bv = bias ? bias[col] : 0.f;
            #pragma unroll
            for (int r = 0; r < 8; ++r) {
                const int row = m0 + wm + i * 16 + khalf * 8 + r;
                float v = acc[i][j][r] + bv;
                if (ACT == 1) v = v > 0.f ? v : 0.f;
                else if (ACT == 2) v = tanhf(v);
                C[(size_t)row * ldc + col] = v;
            }
        }
    }
}

// ---------------------------------------------------------------------------
// GRU recurrence over T steps for a 32-row batch slice. 512 threads = 16 waves.
// xp: [B,T,768] precomputed x@W_ih^T+b_ih (gate order r,z,n). Whh f16 [768,256].
// Each step: hp = h @ Whh^T (WMMA, M=32 N=768 K=256), then fused gates.
// Wave w: M-tile = w&1, N range = (w>>1)*96 .. +96 (6 tiles) -> acc[6] = 48 VGPRs.
// Dynamic LDS: h16[32][256] (16KB) + hp[32][768] f32 (96KB) = 112KB.
// The Whh base is offset by an opaque zero * t so LICM cannot preload/spill the
// loop-invariant weight fragments; they stay as per-step global loads (L2-hot).
// ---------------------------------------------------------------------------
__global__ void __launch_bounds__(512)
gru_kernel(const float* __restrict__ xp, const _Float16* __restrict__ Whh,
           const float* __restrict__ bhh, float* __restrict__ out, int T)
{
    extern __shared__ char smem[];
    _Float16 (*h16)[256] = (_Float16(*)[256])smem;
    float    (*hp)[768]  = (float(*)[768])(smem + 32 * 256 * sizeof(_Float16));

    const int tid     = threadIdx.x;
    const int lane    = tid & 31;
    const int wave    = tid >> 5;            // 0..15
    const int lane_lo = lane & 15;
    const int khalf   = lane >> 4;
    const int b0      = blockIdx.x * 32;

    for (int i = tid; i < 32 * 256; i += 512) ((_Float16*)h16)[i] = (_Float16)0.f;
    float hreg[16];
    #pragma unroll
    for (int i = 0; i < 16; ++i) hreg[i] = 0.f;
    const int bb    = tid >> 4;              // batch row within slice (0..31)
    const int jbase = (tid & 15) * 16;       // hidden-unit range (16 units)
    __syncthreads();

    int opaque_zero;
    asm volatile("s_mov_b32 %0, 0" : "=s"(opaque_zero));

    const int wm  = (wave & 1) * 16;         // M-tile base
    const int wn0 = (wave >> 1) * 96;        // wave owns 6 N-tiles
    #pragma unroll 1
    for (int t = 0; t < T; ++t) {
        // address depends (opaquely) on t -> loads cannot be hoisted out of t-loop
        const _Float16* WhhT = Whh + (size_t)opaque_zero * (size_t)t;
        v8f acc[6] = {};
        #pragma unroll
        for (int kc = 0; kc < 8; ++kc) {
            const int k0 = kc * 32;
            v16h afr;
            {
                const int m = wm + lane_lo;
                h8v lo = *(const h8v*)&h16[m][k0 + khalf * 8];
                h8v hi = *(const h8v*)&h16[m][k0 + 16 + khalf * 8];
                #pragma unroll
                for (int e = 0; e < 8; ++e) { afr[e] = lo[e]; afr[8 + e] = hi[e]; }
            }
            #pragma unroll
            for (int j = 0; j < 6; ++j) {
                const int n = wn0 + j * 16 + lane_lo;
                const _Float16* bp = WhhT + (size_t)n * 256 + k0 + khalf * 8;
                h8v lo = *(const h8v*)bp;
                h8v hi = *(const h8v*)(bp + 16);
                v16h bfr;
                #pragma unroll
                for (int e = 0; e < 8; ++e) { bfr[e] = lo[e]; bfr[8 + e] = hi[e]; }
                acc[j] = __builtin_amdgcn_wmma_f32_16x16x32_f16(
                    false, afr, false, bfr, (short)0, acc[j], false, false);
            }
        }
        // hp -> LDS (+b_hh)
        #pragma unroll
        for (int j = 0; j < 6; ++j) {
            const int col = wn0 + j * 16 + lane_lo;
            const float bv = bhh[col];
            #pragma unroll
            for (int r = 0; r < 8; ++r)
                hp[wm + khalf * 8 + r][col] = acc[j][r] + bv;
        }
        __syncthreads();
        // fused gates: r,z,n  (torch GRU)
        {
            const float* xrow = xp + ((size_t)(b0 + bb) * T + t) * 768;
            float* orow = out + ((size_t)(b0 + bb) * T + t) * 256;
            #pragma unroll
            for (int i = 0; i < 16; ++i) {
                const int j = jbase + i;
                const float r = 1.f / (1.f + __expf(-(xrow[j]       + hp[bb][j])));
                const float z = 1.f / (1.f + __expf(-(xrow[256 + j] + hp[bb][256 + j])));
                const float n = tanhf(xrow[512 + j] + r * hp[bb][512 + j]);
                const float hnew = (1.f - z) * n + z * hreg[i];
                hreg[i] = hnew;
                h16[bb][j] = (_Float16)hnew;
                orow[j] = hnew;
            }
        }
        __syncthreads();
    }
}

// ---------------------------------------------------------------------------
// e[row] = dot(aout[row,:256], w) + b ; one wave per row
// ---------------------------------------------------------------------------
__global__ void attn_e_kernel(const float* __restrict__ aout, const float* __restrict__ w,
                              const float* __restrict__ b, float* __restrict__ e, int rows)
{
    const int lane = threadIdx.x & 31, wave = threadIdx.x >> 5;
    const int row = blockIdx.x * 8 + wave;
    if (row >= rows) return;
    const float* p = aout + (size_t)row * 256;
    float s = 0.f;
    #pragma unroll
    for (int i = 0; i < 8; ++i) s += p[lane + i * 32] * w[lane + i * 32];
    #pragma unroll
    for (int off = 16; off > 0; off >>= 1) s += __shfl_down(s, off, 32);
    if (lane == 0) e[row] = s + b[0];
}

// softmax over T per batch row; block = 64 threads
__global__ void softmax_kernel(const float* __restrict__ e, float* __restrict__ a, int T)
{
    __shared__ float buf[64];
    const int b = blockIdx.x, t = threadIdx.x;
    const float v = e[b * T + t];
    buf[t] = v; __syncthreads();
    for (int s = 32; s > 0; s >>= 1) { if (t < s) buf[t] = fmaxf(buf[t], buf[t + s]); __syncthreads(); }
    const float mx = buf[0]; __syncthreads();
    const float ex = __expf(v - mx);
    buf[t] = ex; __syncthreads();
    for (int s = 32; s > 0; s >>= 1) { if (t < s) buf[t] += buf[t + s]; __syncthreads(); }
    a[b * T + t] = ex / buf[0];
}

// context + concat(demographic): combined[b, 0:256]=sum_t emb*beta*alpha ; [256:272]=demo
__global__ void context_kernel(const float* __restrict__ emb, const float* __restrict__ beta,
                               const float* __restrict__ alpha, const float* __restrict__ demo,
                               float* __restrict__ combined, int T)
{
    const int b = blockIdx.x, h = threadIdx.x;
    float s = 0.f;
    for (int t = 0; t < T; ++t) {
        const size_t idx = ((size_t)b * T + t) * 256 + h;
        s += emb[idx] * beta[idx] * alpha[b * T + t];
    }
    combined[b * 272 + h] = s;
    if (h < 16) combined[b * 272 + 256 + h] = demo[b * 16 + h];
}

// scalar head: out[row*ldo + ldo-1] = dot(h2[row,:256], w) + b ; one wave per row
__global__ void dl_kernel(const float* __restrict__ h2, const float* __restrict__ w,
                          const float* __restrict__ b, float* __restrict__ out, int ldo)
{
    const int lane = threadIdx.x & 31, wave = threadIdx.x >> 5;
    const int row = blockIdx.x * 8 + wave;
    const float* p = h2 + (size_t)row * 256;
    float s = 0.f;
    #pragma unroll
    for (int i = 0; i < 8; ++i) s += p[lane + i * 32] * w[lane + i * 32];
    #pragma unroll
    for (int off = 16; off > 0; off >>= 1) s += __shfl_down(s, off, 32);
    if (lane == 0) out[(size_t)row * ldo + ldo - 1] = s + b[0];
}

// ---------------------------------------------------------------------------
extern "C" void kernel_launch(void* const* d_in, const int* in_sizes, int n_in,
                              void* d_out, int out_size, void* d_ws, size_t ws_size,
                              hipStream_t stream)
{
    (void)in_sizes; (void)n_in; (void)out_size; (void)ws_size;
    constexpr int B = 128, T = 64, C = 8000, H = 256, D = 16, G = 768, M = B * T;

    const float* x       = (const float*)d_in[0];
    const float* demo    = (const float*)d_in[1];
    const float* embed_W = (const float*)d_in[2];
    const float* embed_b = (const float*)d_in[3];
    const float* aW_ih   = (const float*)d_in[4];
    const float* aW_hh   = (const float*)d_in[5];
    const float* ab_ih   = (const float*)d_in[6];
    const float* ab_hh   = (const float*)d_in[7];
    const float* bW_ih   = (const float*)d_in[8];
    const float* bW_hh   = (const float*)d_in[9];
    const float* bb_ih   = (const float*)d_in[10];
    const float* bb_hh   = (const float*)d_in[11];
    const float* attnA_W = (const float*)d_in[12];
    const float* attnA_b = (const float*)d_in[13];
    const float* attnB_W = (const float*)d_in[14];
    const float* attnB_b = (const float*)d_in[15];
    const float* cls_W1  = (const float*)d_in[16];
    const float* cls_b1  = (const float*)d_in[17];
    const float* cls_W2  = (const float*)d_in[18];
    const float* cls_b2  = (const float*)d_in[19];
    const float* dl_W1   = (const float*)d_in[20];
    const float* dl_b1   = (const float*)d_in[21];
    const float* dl_W2   = (const float*)d_in[22];
    const float* dl_b2   = (const float*)d_in[23];
    float* outp = (float*)d_out;

    // workspace partition
    char* ws = (char*)d_ws;
    size_t off = 0;
    auto alloc = [&](size_t bytes) { char* p = ws + off; off += (bytes + 255) & ~(size_t)255; return p; };
    float* embedded = (float*)alloc((size_t)M * H * 4);
    float* xp       = (float*)alloc((size_t)M * G * 4);
    float* gruA     = (float*)alloc((size_t)M * H * 4);
    float* gruB     = (float*)alloc((size_t)M * H * 4);
    float* evec     = (float*)alloc((size_t)M * 4);
    float* alphaw   = (float*)alloc((size_t)M * 4);
    float* betaArr  = (float*)alloc((size_t)M * H * 4);
    float* combined = (float*)alloc((size_t)B * (H + D) * 4);
    float* h1buf    = (float*)alloc((size_t)B * H * 4);
    float* h2buf    = (float*)alloc((size_t)B * H * 4);
    _Float16* embedW16 = (_Float16*)alloc((size_t)C * H * 2);
    _Float16* aWih16   = (_Float16*)alloc((size_t)G * H * 2);
    _Float16* aWhh16   = (_Float16*)alloc((size_t)G * H * 2);
    _Float16* bWih16   = (_Float16*)alloc((size_t)G * H * 2);
    _Float16* bWhh16   = (_Float16*)alloc((size_t)G * H * 2);
    _Float16* attnB16  = (_Float16*)alloc((size_t)H * H * 2);
    _Float16* clsW1_16 = (_Float16*)alloc((size_t)H * (H + D) * 2);
    _Float16* clsW2_16 = (_Float16*)alloc((size_t)C * H * 2);
    _Float16* dlW1_16  = (_Float16*)alloc((size_t)H * (H + D) * 2);

    auto cvt = [&](const float* s, _Float16* d, int n) {
        cvt_f32_f16<<<(n + 255) / 256, 256, 0, stream>>>(s, d, n);
    };
    cvt(embed_W, embedW16, C * H);
    cvt(aW_ih, aWih16, G * H);
    cvt(aW_hh, aWhh16, G * H);
    cvt(bW_ih, bWih16, G * H);
    cvt(bW_hh, bWhh16, G * H);
    cvt(attnB_W, attnB16, H * H);
    cvt(cls_W1, clsW1_16, H * (H + D));
    cvt(cls_W2, clsW2_16, C * H);
    cvt(dl_W1, dlW1_16, H * (H + D));

    // GRU kernel needs 112KB dynamic LDS
    const int gruSmem = 32 * 256 * 2 + 32 * 768 * 4;
    (void)hipFuncSetAttribute(reinterpret_cast<const void*>(gru_kernel),
                              hipFuncAttributeMaxDynamicSharedMemorySize, gruSmem);

    // 1) embedded = x @ embed_W^T + embed_b        [8192,256], K=8000
    gemm_wmma<0><<<dim3(M / 128, H / 64), 256, 0, stream>>>(x, embedW16, embed_b, embedded, M, H, C, H);
    // 2) xp = embedded @ aW_ih^T + ab_ih           [8192,768], K=256
    gemm_wmma<0><<<dim3(M / 128, G / 64), 256, 0, stream>>>(embedded, aWih16, ab_ih, xp, M, G, H, G);
    // 3) GRU alpha
    gru_kernel<<<B / 32, 512, gruSmem, stream>>>(xp, aWhh16, ab_hh, gruA, T);
    // 4) xp = embedded @ bW_ih^T + bb_ih  (reuse buffer)
    gemm_wmma<0><<<dim3(M / 128, G / 64), 256, 0, stream>>>(embedded, bWih16, bb_ih, xp, M, G, H, G);
    // 5) GRU beta
    gru_kernel<<<B / 32, 512, gruSmem, stream>>>(xp, bWhh16, bb_hh, gruB, T);
    // 6) e_alpha + softmax over T
    attn_e_kernel<<<M / 8, 256, 0, stream>>>(gruA, attnA_W, attnA_b, evec, M);
    softmax_kernel<<<B, 64, 0, stream>>>(evec, alphaw, T);
    // 7) beta = tanh(gruB @ attnB_W^T + attnB_b)   [8192,256], K=256
    gemm_wmma<2><<<dim3(M / 128, H / 64), 256, 0, stream>>>(gruB, attnB16, attnB_b, betaArr, M, H, H, H);
    // 8) context + concat demographic
    context_kernel<<<B, H, 0, stream>>>(embedded, betaArr, alphaw, demo, combined, T);
    // 9) h1 = relu(combined @ cls_W1^T + cls_b1)   [128,256], K=272
    gemm_wmma<1><<<dim3(1, H / 64), 256, 0, stream>>>(combined, clsW1_16, cls_b1, h1buf, B, H, H + D, H);
    // 10) logits -> d_out[:, 0:8000], ldc = 8001
    gemm_wmma<0><<<dim3(1, C / 64), 256, 0, stream>>>(h1buf, clsW2_16, cls_b2, outp, B, C, H, C + 1);
    // 11) h2 = relu(combined @ dl_W1^T + dl_b1)
    gemm_wmma<1><<<dim3(1, H / 64), 256, 0, stream>>>(combined, dlW1_16, dl_b1, h2buf, B, H, H + D, H);
    // 12) dl_logit -> d_out[:, 8000]
    dl_kernel<<<B / 8, 256, 0, stream>>>(h2buf, dl_W2, dl_b2, outp, C + 1);
}
